// StoTransformerBlock_75788992906154
// MI455X (gfx1250) — matrix operand order repaired and checked by reading, not compiled, not run on hardware
//
#include <hip/hip_runtime.h>
#include <hip/hip_bf16.h>
#include <math.h>

// ---------------- problem dims (fixed by the reference) ----------------
#define DIM_B   4
#define DIM_H   256
#define DIM_W   128
#define DIM_C   256
#define N_HEADS 8
#define HEAD_D  32
#define WIN     8
#define N_TOK   64      // WIN*WIN
#define HID_DIM 1024
#define M_ROWS  131072  // B*H*W

typedef __attribute__((ext_vector_type(16))) _Float16 v16h;
typedef __attribute__((ext_vector_type(8)))  float    v8f;
typedef __attribute__((ext_vector_type(4)))  unsigned int u32x4;
typedef __attribute__((ext_vector_type(8)))  int          i32x8;
typedef __attribute__((ext_vector_type(4)))  int          i32x4;

#define DEV __device__ __forceinline__

// ---------------- WMMA fragment helpers (wave32, 16x16x32 f16) ----------------
// A fragment: 16(M) x 32(K) tile, row-major in LDS with leading dim `ld`.
// Half-wave group g selects K pairs {0..7,16..23} (g=0) / {8..15,24..31} (g=1).
DEV v16h load_frag_a(const _Float16* s, int ld, int lane) {
  const int g = lane >> 4, m = lane & 15;
  const _Float16* p = s + m * ld;
  v16h a;
#pragma unroll
  for (int i = 0; i < 8; ++i) {
    const int k = (i < 4) ? (8 * g + 2 * i) : (16 + 8 * g + 2 * (i - 4));
    a[2 * i]     = p[k];
    a[2 * i + 1] = p[k + 1];
  }
  return a;
}

// B fragment: 32(K) x 16(N) tile stored TRANSPOSED in LDS as Bt[n][k], ld = row
// stride. Lane group g covers K {8g..8g+7, 16+8g..16+8g+7} (same as A).
DEV v16h load_frag_b(const _Float16* s, int ld, int lane) {
  const int g = lane >> 4, n = lane & 15;
  const _Float16* p = s + n * ld + 8 * g;
  v16h b;
#pragma unroll
  for (int j = 0; j < 8; ++j) { b[j] = p[j]; b[j + 8] = p[j + 16]; }
  return b;
}

DEV v8f wmma16x16x32(v16h a, v16h b, v8f c) {
  return __builtin_amdgcn_wmma_f32_16x16x32_f16(false, a, false, b, (short)0, c,
                                                false, false);
}

DEV float gelu_f(float v) { return 0.5f * v * (1.0f + erff(v * 0.70710678118654752f)); }

DEV unsigned lds_addr_of(const void* p) {
  // flat LDS pointer -> 32-bit LDS byte address (aperture truncation)
  return (unsigned)(unsigned long long)(uintptr_t)p;
}

// ---------------- Tensor Data Mover: 2D tile load, f16 elements ----------------
// Builds D# group0/group1 per CDNA5 ISA §8 and issues TENSOR_LOAD_TO_LDS.
// pad codes: interval 3 -> pad after 16 DWORDs, amount 3 -> 4 DWORDs
// (i.e. 32-f16 rows land with an LDS row stride of 40 f16).
DEV void tdm_load_2d_f16(unsigned lds, unsigned long long gaddr,
                         unsigned tile_d0, unsigned tile_d1,
                         unsigned tensor_d0, unsigned tensor_d1,
                         unsigned long long stride0, bool pad_en) {
  u32x4 g0;
  g0[0] = 1u;                                         // count=1, user descriptor
  g0[1] = lds;                                        // lds_addr [63:32]
  g0[2] = (unsigned)gaddr;                            // global_addr lo
  g0[3] = (unsigned)((gaddr >> 32) & 0x01FFFFFFull)   // global_addr [56:32]
          | (2u << 30);                               // type=2 ("image")
  i32x8 g1;
  unsigned w0 = (1u << 16);                           // data_size code 1 = 2 bytes
  if (pad_en) w0 |= (1u << 20) | (3u << 22) | (3u << 25);
  g1[0] = (int)w0;
  g1[1] = (int)((tensor_d0 & 0xFFFFu) << 16);         // tensor_dim0 [79:48] lo
  g1[2] = (int)(((tensor_d0 >> 16) & 0xFFFFu) |       // tensor_dim0 hi
                ((tensor_d1 & 0xFFFFu) << 16));       // tensor_dim1 lo
  g1[3] = (int)(((tensor_d1 >> 16) & 0xFFFFu) |       // tensor_dim1 hi
                ((tile_d0 & 0xFFFFu) << 16));         // tile_dim0 [127:112]
  g1[4] = (int)(tile_d1 & 0xFFFFu);                   // tile_dim1; tile_dim2=0
  g1[5] = (int)(unsigned)(stride0 & 0xFFFFFFFFull);   // dim0 stride lo32
  g1[6] = (int)(unsigned)((stride0 >> 32) & 0xFFFFu); // dim0 stride hi; dim1 stride=0
  g1[7] = 0;
  const i32x4 z4 = {0, 0, 0, 0};                      // groups 2/3 NULL -> 2D
  const i32x8 z8 = {0, 0, 0, 0, 0, 0, 0, 0};          // unused extra group (6-arg form)
  __builtin_amdgcn_tensor_load_to_lds(g0, g1, z4, z4, z8, 0);
}

// ---------------- LayerNorm (f32 in -> f16 out), one wave per 256-wide row ----------------
__global__ __launch_bounds__(256) void ln_f32_to_f16(
    const float* __restrict__ x, const float* __restrict__ gam,
    const float* __restrict__ bet, _Float16* __restrict__ out, int M) {
  const int lane = threadIdx.x & 31;
  const int wave = threadIdx.x >> 5;
  const int row  = blockIdx.x * 8 + wave;
  if (row >= M) return;
  const float* p = x + (size_t)row * DIM_C + lane * 8;
  float v[8];
  const float4 a4 = *(const float4*)(p);
  const float4 b4 = *(const float4*)(p + 4);
  v[0]=a4.x; v[1]=a4.y; v[2]=a4.z; v[3]=a4.w;
  v[4]=b4.x; v[5]=b4.y; v[6]=b4.z; v[7]=b4.w;
  float s = 0.f;
#pragma unroll
  for (int i = 0; i < 8; ++i) s += v[i];
#pragma unroll
  for (int m = 1; m < 32; m <<= 1) s += __shfl_xor(s, m, 32);
  const float mu = s * (1.0f / DIM_C);
  float q = 0.f;
#pragma unroll
  for (int i = 0; i < 8; ++i) { const float d = v[i] - mu; q += d * d; }
#pragma unroll
  for (int m = 1; m < 32; m <<= 1) q += __shfl_xor(q, m, 32);
  const float inv = rsqrtf(q * (1.0f / DIM_C) + 1e-5f);
  _Float16* o = out + (size_t)row * DIM_C + lane * 8;
#pragma unroll
  for (int i = 0; i < 8; ++i) {
    const int c = lane * 8 + i;
    o[i] = (_Float16)((v[i] - mu) * inv * gam[c] + bet[c]);
  }
}

// ---------------- generic WMMA GEMM: out = epilogue(A(f16)[M,K] @ W(f32)[K,N] + bias) ----------------
// EPI: 0 = f16 store; 1 = GELU+f16; 2 = f32 store; 3 = resid+v, f32 store;
//      4 = f16 store with roll+window-partition row permutation (for q/v).
template <int EPI>
__global__ __launch_bounds__(256) void wmma_gemm(
    const _Float16* __restrict__ A, const float* __restrict__ Wm,
    const float* __restrict__ bias, void* __restrict__ outp,
    const float* __restrict__ resid, const int* __restrict__ hoffp,
    const int* __restrict__ woffp, int M, int N, int K) {
  __shared__ __align__(16) _Float16 As[128][40];  // filled by TDM (padded by DMA)
  __shared__ __align__(16) _Float16 Bs[64][40];   // W tile transposed: Bs[n][k]
  const int tid = threadIdx.x, lane = tid & 31, wave = tid >> 5;
  const int n0 = blockIdx.x * 64;
  const int m0 = blockIdx.y * 128;

  v8f acc[4];
#pragma unroll
  for (int i = 0; i < 4; ++i)
#pragma unroll
    for (int j = 0; j < 8; ++j) acc[i][j] = 0.f;

  for (int k0 = 0; k0 < K; k0 += 32) {
    // A tile 128x32 f16 via Tensor Data Mover (wave 0 issues one descriptor)
    if (wave == 0) {
      tdm_load_2d_f16(lds_addr_of(&As[0][0]),
                      (unsigned long long)(uintptr_t)(A + (size_t)m0 * K + k0),
                      /*tile*/ 32, 128, /*tensor*/ 32, 128,
                      /*stride0*/ (unsigned long long)K, /*pad*/ true);
    }
    // W tile: Bs[n][k] = (f16)Wm[(k0+k)*N + n0+n]; read float4 along n
#pragma unroll
    for (int it = 0; it < 2; ++it) {
      const int c = tid + 256 * it;
      const int k = c >> 4, nq = c & 15;
      const float4 f = *(const float4*)(Wm + (size_t)(k0 + k) * N + n0 + nq * 4);
      Bs[nq * 4 + 0][k] = (_Float16)f.x;
      Bs[nq * 4 + 1][k] = (_Float16)f.y;
      Bs[nq * 4 + 2][k] = (_Float16)f.z;
      Bs[nq * 4 + 3][k] = (_Float16)f.w;
    }
    if (k0 + 32 < K)  // global_prefetch of next W tile
      __builtin_prefetch(Wm + (size_t)(k0 + 32 + (tid >> 3)) * N + n0 + (tid & 7) * 8,
                         0, 1);
    if (wave == 0) __builtin_amdgcn_s_wait_tensorcnt(0);
    __syncthreads();
    const v16h a = load_frag_a(&As[16 * wave][0], 40, lane);
#pragma unroll
    for (int nt = 0; nt < 4; ++nt) {
      const v16h b = load_frag_b(&Bs[16 * nt][0], 40, lane);
      acc[nt] = wmma16x16x32(a, b, acc[nt]);
    }
    __syncthreads();
  }

  const int g = lane >> 4, kn = lane & 15;
  int Hoff = 0, Woff = 0;
  if (EPI == 4) { Hoff = hoffp[0]; Woff = woffp[0]; }
#pragma unroll
  for (int nt = 0; nt < 4; ++nt) {
    const int col = n0 + 16 * nt + kn;
    const float bv = bias[col];
#pragma unroll
    for (int r = 0; r < 8; ++r) {
      const int row = m0 + 16 * wave + 8 * g + r;
      const float v = acc[nt][r] + bv;
      if (EPI == 4) {
        // pixel row -> rolled coords -> window row (window-gathered layout)
        const int b  = row >> 15;
        const int oh = (row >> 7) & (DIM_H - 1);
        const int ow = row & (DIM_W - 1);
        const int rh = (oh + DIM_H - Hoff) & (DIM_H - 1);
        const int rw = (ow + DIM_W - Woff) & (DIM_W - 1);
        const int prow = (((b << 9) + ((rh >> 3) << 4) + (rw >> 3)) << 6) +
                         ((rh & 7) << 3) + (rw & 7);
        ((_Float16*)outp)[(size_t)prow * N + col] = (_Float16)v;
      } else {
        const size_t off = (size_t)row * N + col;
        if (EPI == 0)      ((_Float16*)outp)[off] = (_Float16)v;
        else if (EPI == 1) ((_Float16*)outp)[off] = (_Float16)gelu_f(v);
        else if (EPI == 2) ((float*)outp)[off] = v;
        else               ((float*)outp)[off] = resid[off] + v;
      }
    }
  }
}

// ---------------- shift-mask region id (Swin SW-MSA) ----------------
DEV int region_id(int r, int size, int off) {
  if (off <= 0) return 0;
  if (r < size - WIN) return 0;
  if (r < size - off) return 1;
  return 2;
}

// ---------------- windowed attention: one wave == one head; 4 heads / 128-thread block ----------------
// qg/vg: window-gathered layout [window][token][C] f16 (roll folded in upstream).
// y: same layout.
__global__ __launch_bounds__(128) void win_attn(
    const _Float16* __restrict__ qg, const _Float16* __restrict__ vg,
    const float* __restrict__ rpb, _Float16* __restrict__ y,
    const int* __restrict__ hoffp, const int* __restrict__ woffp) {
  __shared__ __align__(16) _Float16 Qs[4][64][40];  // Q (=K), TDM-padded rows
  __shared__ __align__(16) _Float16 Vt[4][32][72];  // V transposed: Vt[d][token]
  __shared__ __align__(16) _Float16 Ps[4][16][72];  // softmax strip (16 q x 64 k)
  const int Hoff = hoffp[0], Woff = woffp[0];
  const int lane = threadIdx.x & 31, wslot = threadIdx.x >> 5;
  const int wid  = blockIdx.x >> 1;                 // window id 0..2047
  const int head = (blockIdx.x & 1) * 4 + wslot;
  const int wloc = wid & 511;
  const int wh = wloc >> 4, ww = wloc & 15;

  // Q tile (64 tokens x 32 d) via TDM, one descriptor per wave/head
  tdm_load_2d_f16(lds_addr_of(&Qs[wslot][0][0]),
                  (unsigned long long)(uintptr_t)(qg + (size_t)wid * N_TOK * DIM_C +
                                                  head * HEAD_D),
                  /*tile*/ 32, 64, /*tensor*/ 32, 64,
                  /*stride0*/ DIM_C, /*pad*/ true);

  // V tile: contiguous loads + in-register transpose into Vt[d][token]
#pragma unroll
  for (int i = 0; i < 8; ++i) {
    const int chunk = lane + 32 * i;                // 256 = 64 tokens x 4 parts
    const int tok = chunk >> 2, part = chunk & 3;
    const size_t base =
        ((size_t)wid * N_TOK + tok) * DIM_C + head * HEAD_D + part * 8;
    union { uint4 u; _Float16 h[8]; } tv;
    tv.u = *(const uint4*)(vg + base);
#pragma unroll
    for (int j = 0; j < 8; ++j) Vt[wslot][part * 8 + j][tok] = tv.h[j];
  }
  __builtin_amdgcn_s_wait_tensorcnt(0);
  __syncthreads();

  const float scale = 0.17677669529663687f;  // 32^-0.5
  const int g = lane >> 4, kn = lane & 15;

  int kth[4], ktw[4], kcnt[4];
#pragma unroll
  for (int nt = 0; nt < 4; ++nt) {
    const int kt = 16 * nt + kn;
    kth[nt] = kt >> 3; ktw[nt] = kt & 7;
    kcnt[nt] = region_id(wh * 8 + kth[nt], DIM_H, Hoff) * 3 +
               region_id(ww * 8 + ktw[nt], DIM_W, Woff);
  }

  for (int mt = 0; mt < 4; ++mt) {
    // scores strip: Q(16x32) @ K^T(32x64); k=q so Qs serves both A and Bt
    const v16h a = load_frag_a(&Qs[wslot][16 * mt][0], 40, lane);
    v8f sc[4];
#pragma unroll
    for (int nt = 0; nt < 4; ++nt) {
#pragma unroll
      for (int j = 0; j < 8; ++j) sc[nt][j] = 0.f;
      const v16h bfr = load_frag_b(&Qs[wslot][16 * nt][0], 40, lane);
      sc[nt] = wmma16x16x32(a, bfr, sc[nt]);
    }
    // scale + rel-pos bias + shift mask, then row softmax (rows span 16 lanes)
#pragma unroll
    for (int r = 0; r < 8; ++r) {
      const int qt = 16 * mt + 8 * g + r;
      const int qth = qt >> 3, qtw = qt & 7;
      const int qcnt = region_id(wh * 8 + qth, DIM_H, Hoff) * 3 +
                       region_id(ww * 8 + qtw, DIM_W, Woff);
      float e[4];
#pragma unroll
      for (int nt = 0; nt < 4; ++nt) {
        const int idx = (qth - kth[nt] + 7) * 15 + (qtw - ktw[nt] + 7);
        float v = sc[nt][r] * scale + rpb[idx * N_HEADS + head];
        if (qcnt != kcnt[nt]) v -= 100.0f;
        e[nt] = v;
      }
      float mx = fmaxf(fmaxf(e[0], e[1]), fmaxf(e[2], e[3]));
#pragma unroll
      for (int m = 1; m < 16; m <<= 1) mx = fmaxf(mx, __shfl_xor(mx, m, 32));
      float sum = 0.f;
#pragma unroll
      for (int nt = 0; nt < 4; ++nt) { e[nt] = __expf(e[nt] - mx); sum += e[nt]; }
#pragma unroll
      for (int m = 1; m < 16; m <<= 1) sum += __shfl_xor(sum, m, 32);
      const float rs = 1.0f / sum;
#pragma unroll
      for (int nt = 0; nt < 4; ++nt)
        Ps[wslot][8 * g + r][16 * nt + kn] = (_Float16)(e[nt] * rs);
    }
    __syncthreads();
    // O strip = P(16x64) @ V(64x32)
#pragma unroll
    for (int nt2 = 0; nt2 < 2; ++nt2) {
      v8f o;
#pragma unroll
      for (int j = 0; j < 8; ++j) o[j] = 0.f;
#pragma unroll
      for (int ks = 0; ks < 2; ++ks) {
        const v16h pa = load_frag_a(&Ps[wslot][0][32 * ks], 72, lane);
        const v16h vb = load_frag_b(&Vt[wslot][16 * nt2][32 * ks], 72, lane);
        o = wmma16x16x32(pa, vb, o);
      }
#pragma unroll
      for (int r = 0; r < 8; ++r) {
        const int tok = 16 * mt + 8 * g + r;
        y[((size_t)wid * N_TOK + tok) * DIM_C + head * HEAD_D + 16 * nt2 + kn] =
            (_Float16)o[r];
      }
    }
    __syncthreads();
  }
}

// ---------------- window-reverse + inverse roll + residual: x1 = x + unroll(winrev(yp)) ----------------
__global__ __launch_bounds__(256) void scatter_residual(
    const float* __restrict__ x, const float* __restrict__ yp,
    float* __restrict__ x1, const int* __restrict__ hoffp,
    const int* __restrict__ woffp) {
  const int Hoff = hoffp[0], Woff = woffp[0];
  const int idx = blockIdx.x * 256 + threadIdx.x;  // rolled pixel*64 + c-quad
  const int cq = idx & 63;
  const int p  = idx >> 6;
  const int rw = p & (DIM_W - 1);
  const int rh = (p >> 7) & (DIM_H - 1);
  const int b  = p >> 15;
  const int wrow = (b << 9) + ((rh >> 3) << 4) + (rw >> 3);
  const int tok  = ((rh & 7) << 3) + (rw & 7);
  const int oh = (rh + Hoff) & (DIM_H - 1);
  const int ow = (rw + Woff) & (DIM_W - 1);
  const size_t dst = (((size_t)(b * DIM_H + oh)) * DIM_W + ow) * DIM_C + cq * 4;
  const float4 xv = *(const float4*)(x + dst);
  const float4 yv = *(const float4*)(yp + ((size_t)wrow * N_TOK + tok) * DIM_C + cq * 4);
  float4 o;
  o.x = xv.x + yv.x; o.y = xv.y + yv.y; o.z = xv.z + yv.z; o.w = xv.w + yv.w;
  *(float4*)(x1 + dst) = o;
}

// ---------------- depthwise 3x3 SAME + GELU (channels-last) ----------------
__global__ __launch_bounds__(256) void dwconv_gelu(
    const _Float16* __restrict__ t, const float* __restrict__ kw,
    const float* __restrict__ kb, _Float16* __restrict__ t2) {
  const long long idx = (long long)blockIdx.x * 256 + threadIdx.x;
  const int c = (int)(idx & (HID_DIM - 1));
  const long long p = idx >> 10;
  const int w = (int)(p & (DIM_W - 1));
  const int h = (int)((p >> 7) & (DIM_H - 1));
  const int b = (int)(p >> 15);
  float s = 0.f;
#pragma unroll
  for (int dy = 0; dy < 3; ++dy) {
    const int yy = h + dy - 1;
    if (yy < 0 || yy >= DIM_H) continue;
#pragma unroll
    for (int dx = 0; dx < 3; ++dx) {
      const int xx = w + dx - 1;
      if (xx < 0 || xx >= DIM_W) continue;
      s += (float)t[(((size_t)(b * DIM_H + yy)) * DIM_W + xx) * HID_DIM + c] *
           kw[c * 9 + dy * 3 + dx];
    }
  }
  t2[(((size_t)(b * DIM_H + h)) * DIM_W + w) * HID_DIM + c] =
      (_Float16)gelu_f(s + kb[c]);
}

// ---------------- host-side launch chain ----------------
extern "C" void kernel_launch(void* const* d_in, const int* in_sizes, int n_in,
                              void* d_out, int out_size, void* d_ws, size_t ws_size,
                              hipStream_t stream) {
  (void)in_sizes; (void)n_in; (void)out_size; (void)ws_size;
  const float* x     = (const float*)d_in[0];
  const float* n1g   = (const float*)d_in[1];
  const float* n1b   = (const float*)d_in[2];
  const float* wq    = (const float*)d_in[3];
  const float* bq    = (const float*)d_in[4];
  const float* wv    = (const float*)d_in[5];
  const float* bv    = (const float*)d_in[6];
  const float* rpb   = (const float*)d_in[7];
  const float* wproj = (const float*)d_in[8];
  const float* bproj = (const float*)d_in[9];
  const float* n2g   = (const float*)d_in[10];
  const float* n2b   = (const float*)d_in[11];
  const float* w1    = (const float*)d_in[12];
  const float* b1    = (const float*)d_in[13];
  const float* dwk   = (const float*)d_in[14];
  const float* dwb   = (const float*)d_in[15];
  const float* w2    = (const float*)d_in[16];
  const float* b2    = (const float*)d_in[17];
  const int*   hoff  = (const int*)d_in[18];
  const int*   woff  = (const int*)d_in[19];

  char* ws = (char*)d_ws;
  const int M = M_ROWS;
  _Float16* xn  = (_Float16*)(ws);                 //  64 MB  LN output (reused for LN2)
  _Float16* qh  = (_Float16*)(ws + 67108864ull);   //  64 MB  q, window-gathered f16
  _Float16* vh  = (_Float16*)(ws + 134217728ull);  //  64 MB  v, window-gathered f16
  _Float16* yh  = (_Float16*)(ws + 201326592ull);  //  64 MB  attn out (window order)
  float*    yp  = (float*)   (ws + 268435456ull);  // 128 MB  proj out f32
  float*    x1  = (float*)   (ws + 402653184ull);  // 128 MB  x + attn (f32)
  _Float16* th  = (_Float16*)(ws + 536870912ull);  // 256 MB  w1+gelu out f16
  _Float16* t2h = (_Float16*)(ws + 805306368ull);  // 256 MB  dwconv+gelu out f16
  float*    out = (float*)d_out;

  ln_f32_to_f16<<<M / 8, 256, 0, stream>>>(x, n1g, n1b, xn, M);
  wmma_gemm<4><<<dim3(4, M / 128), 256, 0, stream>>>(xn, wq, bq, qh, nullptr, hoff, woff, M, 256, 256);
  wmma_gemm<4><<<dim3(4, M / 128), 256, 0, stream>>>(xn, wv, bv, vh, nullptr, hoff, woff, M, 256, 256);
  win_attn<<<4096, 128, 0, stream>>>(qh, vh, rpb, yh, hoff, woff);
  wmma_gemm<2><<<dim3(4, M / 128), 256, 0, stream>>>(yh, wproj, bproj, yp, nullptr, nullptr, nullptr, M, 256, 256);
  scatter_residual<<<(M * 64) / 256, 256, 0, stream>>>(x, yp, x1, hoff, woff);
  ln_f32_to_f16<<<M / 8, 256, 0, stream>>>(x1, n2g, n2b, xn, M);
  wmma_gemm<1><<<dim3(16, M / 128), 256, 0, stream>>>(xn, w1, b1, th, nullptr, nullptr, nullptr, M, 1024, 256);
  dwconv_gelu<<<(M / 256) * 1024, 256, 0, stream>>>(th, dwk, dwb, t2h);
  wmma_gemm<3><<<dim3(4, M / 128), 256, 0, stream>>>(t2h, w2, b2, out, x1, nullptr, nullptr, M, 256, 1024);
}
